// MRNNCell_27719718928914
// MI455X (gfx1250) — compile-verified, matrix operand order checked by reading, not run on hardware
//
#include <hip/hip_runtime.h>

#define T_STEPS   40
#define BATCH     2048
#define UNITS     1024
#define INPUT_DIM 1024
#define QB        8
#define DDIM      (UNITS + INPUT_DIM)   // 2048

#define BM 128
#define BN 128
#define BK 32

typedef __attribute__((ext_vector_type(16))) __bf16 v16bf;
typedef __attribute__((ext_vector_type(8)))  __bf16 v8bf;
typedef __attribute__((ext_vector_type(8)))  float  v8f;

// ---------------------------------------------------------------------------
// CDNA5 async global->LDS copy (ASYNCcnt path).  INST_OFFSET is added to both
// the LDS and the global address, so one base VGPR each + immediate suffices.
// ---------------------------------------------------------------------------
template <int OFF>
__device__ __forceinline__ void async_b128(unsigned lds_addr, const void* gaddr)
{
    asm volatile("global_load_async_to_lds_b128 %0, %1, off offset:%2"
                 :: "v"(lds_addr), "v"(gaddr), "n"(OFF) : "memory");
}
__device__ __forceinline__ void wait_async0()
{
    asm volatile("s_wait_asynccnt 0" ::: "memory");
}
__device__ __forceinline__ unsigned lds_addr_of(const void* p)
{
    // generic LDS pointer: low 32 bits are the 0-based LDS offset
    return (unsigned)(unsigned long long)p;
}

// ---------------------------------------------------------------------------
// Kernel 1: synthesize all 40 per-step weight matrices in bf16, TRANSPOSED:
//   Qw viewed as (D, U, QB)  ->  W out as (T, U, D) bf16  (W^T per step).
// ---------------------------------------------------------------------------
#define TCHUNK 8
__global__ __launch_bounds__(1024) void build_w(const float* __restrict__ Qw,
                                                const float* __restrict__ Bt,
                                                __bf16* __restrict__ W)
{
    __shared__ float  sBt[T_STEPS * QB];
    __shared__ __bf16 sT[TCHUNK][32 * 33];

    const int tid = threadIdx.x;
    if (tid < T_STEPS * QB) sBt[tid] = Bt[tid];

    const int ux = tid & 31;
    const int dy = tid >> 5;
    const int u0 = blockIdx.x * 32;
    const int d0 = blockIdx.y * 32;

    const float4* q4 = (const float4*)(Qw + ((long)(d0 + dy) * UNITS + (u0 + ux)) * QB);
    const float4 qa = q4[0];
    const float4 qb = q4[1];
    __syncthreads();

    for (int t0 = 0; t0 < T_STEPS; t0 += TCHUNK) {
        #pragma unroll
        for (int tt = 0; tt < TCHUNK; ++tt) {
            const float* bt = &sBt[(t0 + tt) * QB];
            float w = qa.x * bt[0] + qa.y * bt[1] + qa.z * bt[2] + qa.w * bt[3]
                    + qb.x * bt[4] + qb.y * bt[5] + qb.z * bt[6] + qb.w * bt[7];
            sT[tt][ux * 33 + dy] = (__bf16)w;
        }
        __syncthreads();
        const int dx = tid & 31;
        const int uy = tid >> 5;
        #pragma unroll
        for (int tt = 0; tt < TCHUNK; ++tt) {
            W[(long)(t0 + tt) * UNITS * DDIM + (long)(u0 + uy) * DDIM + (d0 + dx)] =
                sT[tt][uy * 33 + dx];
        }
        __syncthreads();
    }
}

// ---------------------------------------------------------------------------
// Kernel 2: convert fp32 h0 -> bf16 state buffer.
// ---------------------------------------------------------------------------
__global__ __launch_bounds__(256) void cvt_h0(const float* __restrict__ h0,
                                              __bf16* __restrict__ h)
{
    int i = (blockIdx.x * 256 + threadIdx.x) * 4;
    float4 f = *(const float4*)(h0 + i);
    h[i + 0] = (__bf16)f.x;
    h[i + 1] = (__bf16)f.y;
    h[i + 2] = (__bf16)f.z;
    h[i + 3] = (__bf16)f.w;
}

// ---------------------------------------------------------------------------
// Kernel 3: one recurrence step.
//   128x128 C-tile / 128 threads (4 waves, 64x64 each = 16 WMMA acc).
//   ALL global->LDS movement is async (no VGPR staging -> no spills),
//   double-buffered LDS, one s_wait_asynccnt + one barrier per K-tile.
//   64 K-tiles: it 0..31 read bf16 h_prev, it 32..63 read fp32 x (cvt at
//   fragment build).  W is (T,U,D) so B rows are K-contiguous.
// ---------------------------------------------------------------------------
__global__ __launch_bounds__(128) void mrnn_step(const __bf16* __restrict__ hprev, // (B,U)
                                                 const float*  __restrict__ x,     // (B,T,I)
                                                 const __bf16* __restrict__ W,     // (T,U,D)
                                                 const float*  __restrict__ bias,  // (U)
                                                 float*        __restrict__ out,   // (B,T,U)
                                                 __bf16*       __restrict__ hnext, // (B,U)
                                                 int t)
{
    __shared__ __bf16 sB[2][BN * BK];          // 2 x 8 KB
    __shared__ char   sA[2][BM * BK * 4];      // 2 x 16 KB: bf16 (h) or fp32 (x)

    const int tid  = threadIdx.x;
    const int lane = tid & 31;
    const int wave = tid >> 5;
    const int wm   = wave >> 1;
    const int wn   = wave & 1;
    const int m0   = blockIdx.x * BM;
    const int n0   = blockIdx.y * BN;
    const int half = lane >> 4;
    const int l15  = lane & 15;

    const __bf16* Wt   = W + (long)t * UNITS * DDIM;
    const __bf16* Arow = hprev + (long)(m0 + tid) * UNITS;
    const float*  Xrow = x + (long)(m0 + tid) * T_STEPS * INPUT_DIM + (long)t * INPUT_DIM;
    const __bf16* Brow = Wt + (long)(n0 + tid) * DDIM;

    v8f acc[4][4];
    #pragma unroll
    for (int i = 0; i < 4; ++i)
        #pragma unroll
        for (int j = 0; j < 4; ++j)
            acc[i][j] = (v8f){0.f, 0.f, 0.f, 0.f, 0.f, 0.f, 0.f, 0.f};

    auto issueB = [&](int it) {
        const int buf = it & 1;
        unsigned l = lds_addr_of(&sB[buf][tid * BK]);
        const void* g = (const void*)(Brow + it * BK);
        async_b128<0>(l, g);  async_b128<16>(l, g);
        async_b128<32>(l, g); async_b128<48>(l, g);
    };
    auto issueAh = [&](int it) {           // bf16 h tile: 64B per row
        const int buf = it & 1;
        unsigned l = lds_addr_of(&sA[buf][tid * BK * 2]);
        const void* g = (const void*)(Arow + it * BK);
        async_b128<0>(l, g);  async_b128<16>(l, g);
        async_b128<32>(l, g); async_b128<48>(l, g);
    };
    auto issueAx = [&](int it) {           // fp32 x tile: 128B per row
        const int buf = it & 1;
        unsigned l = lds_addr_of(&sA[buf][tid * BK * 4]);
        const void* g = (const void*)(Xrow + (it - 32) * BK);
        async_b128<0>(l, g);   async_b128<16>(l, g);
        async_b128<32>(l, g);  async_b128<48>(l, g);
        async_b128<64>(l, g);  async_b128<80>(l, g);
        async_b128<96>(l, g);  async_b128<112>(l, g);
    };

    auto loadBfrag = [&](int buf, v16bf* bfr) {
        #pragma unroll
        for (int j = 0; j < 4; ++j) {
            const int col = wn * 64 + j * 16 + l15;
            const __bf16* base = &sB[buf][col * BK];
            const v8bf p0 = *(const v8bf*)(base + half * 8);
            const v8bf p1 = *(const v8bf*)(base + 16 + half * 8);
            #pragma unroll
            for (int e = 0; e < 8; ++e) { bfr[j][e] = p0[e]; bfr[j][8 + e] = p1[e]; }
        }
    };
    auto domma = [&](const v16bf* af, const v16bf* bfr) {
        #pragma unroll
        for (int i = 0; i < 4; ++i)
            #pragma unroll
            for (int j = 0; j < 4; ++j)
                acc[i][j] = __builtin_amdgcn_wmma_f32_16x16x32_bf16(
                    false, af[i], false, bfr[j], (short)0, acc[i][j], false, false);
    };

    // ---- h phase: tiles 0..31 ----
    issueAh(0); issueB(0);
    for (int it = 0; it < 32; ++it) {
        wait_async0();
        __syncthreads();
        if (it + 1 < 32) { issueAh(it + 1); issueB(it + 1); }
        else             { issueAx(32);     issueB(32);     }
        const int buf = it & 1;
        v16bf af[4], bfr[4];
        #pragma unroll
        for (int i = 0; i < 4; ++i) {
            const int row = wm * 64 + i * 16 + l15;
            const __bf16* base = (const __bf16*)&sA[buf][0] + row * BK;
            const v8bf p0 = *(const v8bf*)(base + half * 8);
            const v8bf p1 = *(const v8bf*)(base + 16 + half * 8);
            #pragma unroll
            for (int e = 0; e < 8; ++e) { af[i][e] = p0[e]; af[i][8 + e] = p1[e]; }
        }
        loadBfrag(buf, bfr);
        domma(af, bfr);
    }
    // ---- x phase: tiles 32..63 ----
    for (int it = 32; it < 64; ++it) {
        wait_async0();
        __syncthreads();
        if (it + 1 < 64) { issueAx(it + 1); issueB(it + 1); }
        const int buf = it & 1;
        v16bf af[4], bfr[4];
        #pragma unroll
        for (int i = 0; i < 4; ++i) {
            const int row = wm * 64 + i * 16 + l15;
            const float* base = (const float*)&sA[buf][0] + row * BK;
            const float4 f0 = *(const float4*)(base + half * 8);
            const float4 f1 = *(const float4*)(base + half * 8 + 4);
            const float4 f2 = *(const float4*)(base + 16 + half * 8);
            const float4 f3 = *(const float4*)(base + 16 + half * 8 + 4);
            af[i][0]  = (__bf16)f0.x; af[i][1]  = (__bf16)f0.y;
            af[i][2]  = (__bf16)f0.z; af[i][3]  = (__bf16)f0.w;
            af[i][4]  = (__bf16)f1.x; af[i][5]  = (__bf16)f1.y;
            af[i][6]  = (__bf16)f1.z; af[i][7]  = (__bf16)f1.w;
            af[i][8]  = (__bf16)f2.x; af[i][9]  = (__bf16)f2.y;
            af[i][10] = (__bf16)f2.z; af[i][11] = (__bf16)f2.w;
            af[i][12] = (__bf16)f3.x; af[i][13] = (__bf16)f3.y;
            af[i][14] = (__bf16)f3.z; af[i][15] = (__bf16)f3.w;
        }
        loadBfrag(buf, bfr);
        domma(af, bfr);
    }

    // ---- epilogue: bias + tanh, write fp32 output and bf16 next-state ----
    #pragma unroll
    for (int j = 0; j < 4; ++j) {
        const int n = n0 + wn * 64 + j * 16 + l15;
        const float bv = bias[n];
        #pragma unroll
        for (int i = 0; i < 4; ++i) {
            #pragma unroll
            for (int r = 0; r < 8; ++r) {
                const int m = m0 + wm * 64 + i * 16 + half * 8 + r;
                const float h = tanhf(acc[i][j][r] + bv);
                out[(long)m * T_STEPS * UNITS + (long)t * UNITS + n] = h;
                hnext[(long)m * UNITS + n] = (__bf16)h;
            }
        }
    }
}

// ---------------------------------------------------------------------------
extern "C" void kernel_launch(void* const* d_in, const int* in_sizes, int n_in,
                              void* d_out, int out_size, void* d_ws, size_t ws_size,
                              hipStream_t stream)
{
    (void)in_sizes; (void)n_in; (void)out_size; (void)ws_size;

    const float* x    = (const float*)d_in[0];
    const float* h0   = (const float*)d_in[1];
    const float* Qw   = (const float*)d_in[2];
    const float* bias = (const float*)d_in[3];
    const float* Bt   = (const float*)d_in[4];
    float* out = (float*)d_out;

    __bf16* wsW  = (__bf16*)d_ws;                            // (T,U,D) bf16
    __bf16* wsH0 = wsW + (size_t)T_STEPS * UNITS * DDIM;
    __bf16* wsH1 = wsH0 + (size_t)BATCH * UNITS;

    build_w<<<dim3(UNITS / 32, DDIM / 32), 1024, 0, stream>>>(Qw, Bt, wsW);
    cvt_h0<<<(BATCH * UNITS) / (256 * 4), 256, 0, stream>>>(h0, wsH0);

    for (int t = 0; t < T_STEPS; ++t) {
        __bf16* hp = (t & 1) ? wsH1 : wsH0;
        __bf16* hn = (t & 1) ? wsH0 : wsH1;
        mrnn_step<<<dim3(BATCH / BM, UNITS / BN), 128, 0, stream>>>(
            hp, x, wsW, bias, out, hn, t);
    }
}
